// PointNet2_12472585027596
// MI455X (gfx1250) — compile-verified, matrix operand order checked by reading, not run on hardware
//
#include <hip/hip_runtime.h>
#include <hip/hip_bf16.h>

// ============================================================================
// PointNet++ forward for MI455X (gfx1250, wave32, WMMA).
// All dense layers go through one WMMA GEMM kernel:
//   Y[M,Cout] = X[M,Kp(f32,zero-padded)] * Wp[CoutP,Kp(f16,pre-padded)]^T + b
// using v_wmma_f32_16x16x32_f16 (f16 in, f32 accumulate).
// Invariants guaranteed by the launcher: M % 64 == 0, Kp % 32 == 0,
// CoutP % 32 == 0  -> the GEMM hot loop has no bounds checks / branches.
// ============================================================================

typedef __attribute__((ext_vector_type(16))) _Float16 v16h;
typedef __attribute__((ext_vector_type(8)))  _Float16 v8h;
typedef __attribute__((ext_vector_type(8)))  float    v8f;

#define BNEPS 1e-5f

__device__ __forceinline__ v8h pack8(float4 a, float4 b) {
  v8h h;
  h[0] = (_Float16)a.x; h[1] = (_Float16)a.y; h[2] = (_Float16)a.z; h[3] = (_Float16)a.w;
  h[4] = (_Float16)b.x; h[5] = (_Float16)b.y; h[6] = (_Float16)b.z; h[7] = (_Float16)b.w;
  return h;
}
__device__ __forceinline__ v16h cat16(v8h lo, v8h hi) {
  return __builtin_shufflevector(lo, hi, 0,1,2,3,4,5,6,7,8,9,10,11,12,13,14,15);
}

// ---------------------------------------------------------------------------
// Farthest point sampling: one block per batch, 512 threads, dist in registers
// ---------------------------------------------------------------------------
__global__ __launch_bounds__(512) void k_fps(const float* __restrict__ xyz,
                                             int n, int stride, int npoint,
                                             int* __restrict__ fid) {
  const int b = blockIdx.x, tid = threadIdx.x;
  const float* P = xyz + (size_t)b * n * stride;
  const int per = (n + 511) >> 9;   // <= 32
  float dl[32];
#pragma unroll
  for (int i = 0; i < 32; ++i) dl[i] = 1e10f;
  __shared__ float sv[512];
  __shared__ int   si[512];
  int far = 0;
  if (tid == 0) fid[b * npoint] = 0;
  for (int it = 1; it < npoint; ++it) {
    const float cx = P[(size_t)far * stride + 0];
    const float cy = P[(size_t)far * stride + 1];
    const float cz = P[(size_t)far * stride + 2];
    float best = -1.0f; int bj = 0;
    for (int i = 0; i < per; ++i) {
      int j = tid + (i << 9);
      if (j < n) {
        float dx = P[(size_t)j * stride + 0] - cx;
        float dy = P[(size_t)j * stride + 1] - cy;
        float dz = P[(size_t)j * stride + 2] - cz;
        float d = dx * dx + dy * dy + dz * dz;
        if (d < dl[i]) dl[i] = d;
        if (dl[i] > best) { best = dl[i]; bj = j; }
      }
    }
    sv[tid] = best; si[tid] = bj;
    __syncthreads();
    for (int s = 256; s > 0; s >>= 1) {
      if (tid < s && sv[tid + s] > sv[tid]) { sv[tid] = sv[tid + s]; si[tid] = si[tid + s]; }
      __syncthreads();
    }
    far = si[0];
    __syncthreads();
    if (tid == 0) fid[b * npoint + it] = far;
  }
}

// ---------------------------------------------------------------------------
__global__ void k_gather(const float* __restrict__ xyz, int stride, int n,
                         const int* __restrict__ fid, int np,
                         float* __restrict__ out, int total) {
  int t = blockIdx.x * blockDim.x + threadIdx.x;
  if (t >= total) return;
  int b = t / np;
  int g = fid[t];
  const float* p = xyz + ((size_t)b * n + g) * stride;
  out[t * 3 + 0] = p[0]; out[t * 3 + 1] = p[1]; out[t * 3 + 2] = p[2];
}

// ---------------------------------------------------------------------------
// Ball query: first nsample points (in index order) within radius, pad w/first
// ---------------------------------------------------------------------------
__global__ void k_ball(const float* __restrict__ xyz, int n, int stride,
                       const float* __restrict__ nxyz, int np, int ns, float r2,
                       int* __restrict__ idx, int total) {
  int t = blockIdx.x * blockDim.x + threadIdx.x;
  if (t >= total) return;
  int b = t / np;
  const float* P = xyz + (size_t)b * n * stride;
  float cx = nxyz[t * 3 + 0], cy = nxyz[t * 3 + 1], cz = nxyz[t * 3 + 2];
  int cnt = 0, first = 0; bool has = false;
  for (int j = 0; j < n; ++j) {
    float dx = P[(size_t)j * stride + 0] - cx;
    float dy = P[(size_t)j * stride + 1] - cy;
    float dz = P[(size_t)j * stride + 2] - cz;
    float d = dx * dx + dy * dy + dz * dz;
    if (d <= r2) {
      if (!has) { first = j; has = true; }
      idx[(size_t)t * ns + cnt] = j;
      if (++cnt == ns) break;
    }
  }
  for (; cnt < ns; ++cnt) idx[(size_t)t * ns + cnt] = first;
}

// ---------------------------------------------------------------------------
// Build grouped input rows: [gxyz - center (3) | point feats (cp) | pad]
// ---------------------------------------------------------------------------
__global__ void k_group(const float* __restrict__ xyz, int stride, int n,
                        const float* __restrict__ pts, int cp,
                        const float* __restrict__ nxyz, const int* __restrict__ idx,
                        int np, int ns, float* __restrict__ X, int Kp, int total) {
  int t = blockIdx.x * blockDim.x + threadIdx.x;
  if (t >= total) return;
  int q = t / ns;
  int s = t - q * ns;
  int b = q / np;
  int g = idx[(size_t)q * ns + s];
  const float* P = xyz + ((size_t)b * n + g) * stride;
  const float* c0 = nxyz + (size_t)q * 3;
  float* row = X + (size_t)t * Kp;
  row[0] = P[0] - c0[0]; row[1] = P[1] - c0[1]; row[2] = P[2] - c0[2];
  const float* F = pts + ((size_t)b * n + g) * cp;
  for (int c = 0; c < cp; ++c) row[3 + c] = F[c];
  for (int k = 3 + cp; k < Kp; ++k) row[k] = 0.0f;
}

__global__ void k_group_all(const float* __restrict__ x2,
                            const float* __restrict__ pts, int cp,
                            float* __restrict__ X, int Kp, int total) {
  int t = blockIdx.x * blockDim.x + threadIdx.x;
  if (t >= total) return;
  float* row = X + (size_t)t * Kp;
  row[0] = x2[t * 3 + 0]; row[1] = x2[t * 3 + 1]; row[2] = x2[t * 3 + 2];
  for (int c = 0; c < cp; ++c) row[3 + c] = pts[(size_t)t * cp + c];
  for (int k = 3 + cp; k < Kp; ++k) row[k] = 0.0f;
}

// ---------------------------------------------------------------------------
// Pre-pad + f32->f16 convert weights: W[Cout,K] -> Wp[CoutP,Kp] (zeros outside)
// ---------------------------------------------------------------------------
__global__ void k_pad_w(const float* __restrict__ W, int K, int Cout,
                        _Float16* __restrict__ Wp, int Kp, int total) {
  int t = blockIdx.x * blockDim.x + threadIdx.x;
  if (t >= total) return;
  int c = t / Kp, k = t - c * Kp;
  Wp[t] = (c < Cout && k < K) ? (_Float16)W[(size_t)c * K + k] : (_Float16)0.0f;
}

// ---------------------------------------------------------------------------
// WMMA GEMM: block = 128 threads (4 waves), block tile 64(M) x 32(Cout).
// Each wave owns one 16-row strip and two 16x16 N-tiles (2 accumulators).
// A: f32 global -> f16 LDS (4x b128 loads + 2x ds_store_b128 per thread/step).
// B: f16 pre-padded global, fragments loaded directly (global_load_b128),
//    cached in WGP$/L2 and reused by every M-block.
// No bounds checks in the K loop (M%64==0, Kp%32==0, CoutP%32==0).
// ---------------------------------------------------------------------------
__global__ __launch_bounds__(128) void k_gemm(const float* __restrict__ X,
                                              const _Float16* __restrict__ Wp,
                                              const float* __restrict__ bias,
                                              float* __restrict__ Y, int ldy,
                                              int Kp, int Cout) {
  __shared__ alignas(16) _Float16 As[64 * 40];   // row stride 40 halves (80B)
  const int tid  = threadIdx.x;
  const int wave = tid >> 5;
  const int lane = tid & 31;
  const int half = lane >> 4;        // 0: K 0-7/16-23, 1: K 8-15/24-31
  const int mrow = lane & 15;
  const int bM = blockIdx.x * 64;
  const int bN = blockIdx.y * 32;

  const int ar = tid >> 1;           // staging row 0..63
  const int ac = (tid & 1) * 16;     // staging chunk 0 or 16
  const float*    Xr  = X + (size_t)(bM + ar) * Kp + ac;
  _Float16*       dst = &As[ar * 40 + ac];
  const _Float16* W0  = Wp + (size_t)(bN + mrow) * Kp + half * 8;
  const _Float16* W1  = Wp + (size_t)(bN + 16 + mrow) * Kp + half * 8;
  const int arow = (wave * 16 + mrow) * 40 + half * 8;

  v8f acc0 = {}, acc1 = {};
  for (int kb = 0; kb < Kp; kb += 32) {
    const float4* src = (const float4*)(Xr + kb);
    float4 f0 = src[0], f1 = src[1], f2 = src[2], f3 = src[3];
    *(v8h*)(dst + 0) = pack8(f0, f1);
    *(v8h*)(dst + 8) = pack8(f2, f3);
    __syncthreads();
    v8h alo = *(const v8h*)&As[arow];
    v8h ahi = *(const v8h*)&As[arow + 16];
    v16h a = cat16(alo, ahi);
    v16h b0 = cat16(*(const v8h*)(W0 + kb), *(const v8h*)(W0 + kb + 16));
    v16h b1 = cat16(*(const v8h*)(W1 + kb), *(const v8h*)(W1 + kb + 16));
    acc0 = __builtin_amdgcn_wmma_f32_16x16x32_f16(false, a, false, b0,
                                                  (short)0, acc0, false, false);
    acc1 = __builtin_amdgcn_wmma_f32_16x16x32_f16(false, a, false, b1,
                                                  (short)0, acc1, false, false);
    __syncthreads();
  }
  // C/D layout: VGPR r -> M = 8*half + r, N = lane&15
  const int col0 = bN + mrow;
  const int col1 = bN + 16 + mrow;
#pragma unroll
  for (int r = 0; r < 8; ++r) {
    int row = bM + wave * 16 + half * 8 + r;
    if (col0 < Cout) Y[(size_t)row * ldy + col0] = acc0[r] + bias[col0];
    if (col1 < Cout) Y[(size_t)row * ldy + col1] = acc1[r] + bias[col1];
  }
}

// ---------------------------------------------------------------------------
// Single-pass BatchNorm stats: each block scans 64 contiguous rows, LDS
// ds_add_f32 per-channel accumulation, then one global atomic per channel.
// ---------------------------------------------------------------------------
__global__ void k_zero(float* __restrict__ p, int n) {
  int t = blockIdx.x * blockDim.x + threadIdx.x;
  if (t < n) p[t] = 0.0f;
}

__global__ __launch_bounds__(256) void k_bn_stats(const float* __restrict__ Y,
                                                  long long total, int C,
                                                  float* __restrict__ sum,
                                                  float* __restrict__ sumsq) {
  __shared__ float ls[1024], lq[1024];
  const int tid = threadIdx.x;
  for (int c = tid; c < C; c += 256) { ls[c] = 0.0f; lq[c] = 0.0f; }
  __syncthreads();
  long long base = (long long)blockIdx.x * (64LL * C);
  long long end  = base + 64LL * C;
  if (end > total) end = total;
  for (long long i = base + tid; i < end; i += 256) {
    float v = Y[i];
    int c = (int)(i % C);
    atomicAdd(&ls[c], v);
    atomicAdd(&lq[c], v * v);
  }
  __syncthreads();
  for (int c = tid; c < C; c += 256) {
    atomicAdd(&sum[c],   ls[c]);
    atomicAdd(&sumsq[c], lq[c]);
  }
}

__global__ void k_bn_apply(float* __restrict__ Y, long long total, int C, float invM,
                           const float* __restrict__ sum, const float* __restrict__ sumsq,
                           const float* __restrict__ g, const float* __restrict__ beta) {
  long long i = (long long)blockIdx.x * blockDim.x + threadIdx.x;
  if (i >= total) return;
  int c = (int)(i % C);
  float m = sum[c] * invM;
  float v = fmaxf(sumsq[c] * invM - m * m, 0.0f);
  float x = Y[i];
  Y[i] = fmaxf((x - m) * rsqrtf(v + BNEPS) * g[c] + beta[c], 0.0f);
}

// ---------------------------------------------------------------------------
__global__ void k_maxpool(const float* __restrict__ Y, int ns, int C,
                          float* __restrict__ out, int total) {
  int t = blockIdx.x * blockDim.x + threadIdx.x;
  if (t >= total) return;
  int c = t % C;
  int grp = t / C;
  const float* p = Y + (size_t)grp * ns * C + c;
  float m = -3.0e38f;
  for (int s = 0; s < ns; ++s) m = fmaxf(m, p[(size_t)s * C]);
  out[t] = m;
}

// ---------------------------------------------------------------------------
// Feature-propagation input builders
// ---------------------------------------------------------------------------
__global__ void k_fp3_build(const float* __restrict__ l2p,
                            const float* __restrict__ l3p,
                            float* __restrict__ X, int total) {
  int t = blockIdx.x * blockDim.x + threadIdx.x;
  if (t >= total) return;
  int b = t >> 7;                               // / 128
  float* row = X + (size_t)t * 1280;
  for (int c = 0; c < 256; ++c)  row[c]       = l2p[(size_t)t * 256 + c];
  for (int c = 0; c < 1024; ++c) row[256 + c] = l3p[(size_t)b * 1024 + c];
}

__global__ void k_fp_build(const float* __restrict__ xyz1,
                           const float* __restrict__ xyz2, int n1, int n2,
                           const float* __restrict__ p1, int c1,
                           const float* __restrict__ p2, int c2,
                           float* __restrict__ X, int Kp, int total) {
  int t = blockIdx.x * blockDim.x + threadIdx.x;
  if (t >= total) return;
  int b = t / n1;
  float px = xyz1[t * 3 + 0], py = xyz1[t * 3 + 1], pz = xyz1[t * 3 + 2];
  const float* Q = xyz2 + (size_t)b * n2 * 3;
  float d0 = 3e38f, d1 = 3e38f, d2 = 3e38f;
  int   i0 = 0, i1 = 0, i2 = 0;
  for (int j = 0; j < n2; ++j) {
    float dx = Q[j * 3 + 0] - px, dy = Q[j * 3 + 1] - py, dz = Q[j * 3 + 2] - pz;
    float d = dx * dx + dy * dy + dz * dz;
    if (d < d0)      { d2 = d1; i2 = i1; d1 = d0; i1 = i0; d0 = d; i0 = j; }
    else if (d < d1) { d2 = d1; i2 = i1; d1 = d;  i1 = j; }
    else if (d < d2) { d2 = d;  i2 = j; }
  }
  float w0 = 1.0f / (d0 + 1e-8f), w1 = 1.0f / (d1 + 1e-8f), w2 = 1.0f / (d2 + 1e-8f);
  float ws = w0 + w1 + w2; w0 /= ws; w1 /= ws; w2 /= ws;
  float* row = X + (size_t)t * Kp;
  for (int c = 0; c < c1; ++c) row[c] = p1[(size_t)t * c1 + c];
  const float* q0 = p2 + ((size_t)b * n2 + i0) * c2;
  const float* q1 = p2 + ((size_t)b * n2 + i1) * c2;
  const float* q2 = p2 + ((size_t)b * n2 + i2) * c2;
  for (int c = 0; c < c2; ++c)
    row[c1 + c] = w0 * q0[c] + w1 * q1[c] + w2 * q2[c];
  for (int k = c1 + c2; k < Kp; ++k) row[k] = 0.0f;
}

__global__ void k_fp1_build(const float* __restrict__ xyz6,
                            const float* __restrict__ l1xyz,
                            const float* __restrict__ fp2o,
                            float* __restrict__ X, int total) {
  int t = blockIdx.x * blockDim.x + threadIdx.x;
  if (t >= total) return;
  int b = t >> 14;                              // / 16384
  const float* pt = xyz6 + (size_t)t * 6;
  const float* Q  = l1xyz + (size_t)b * 512 * 3;
  float px = pt[0], py = pt[1], pz = pt[2];
  float d0 = 3e38f, d1 = 3e38f, d2 = 3e38f;
  int   i0 = 0, i1 = 0, i2 = 0;
  for (int j = 0; j < 512; ++j) {
    float dx = Q[j * 3 + 0] - px, dy = Q[j * 3 + 1] - py, dz = Q[j * 3 + 2] - pz;
    float d = dx * dx + dy * dy + dz * dz;
    if (d < d0)      { d2 = d1; i2 = i1; d1 = d0; i1 = i0; d0 = d; i0 = j; }
    else if (d < d1) { d2 = d1; i2 = i1; d1 = d;  i1 = j; }
    else if (d < d2) { d2 = d;  i2 = j; }
  }
  float w0 = 1.0f / (d0 + 1e-8f), w1 = 1.0f / (d1 + 1e-8f), w2 = 1.0f / (d2 + 1e-8f);
  float ws = w0 + w1 + w2; w0 /= ws; w1 /= ws; w2 /= ws;
  float* row = X + (size_t)t * 160;
  row[0] = pt[0]; row[1] = pt[1]; row[2] = pt[2];       // l0_xyz
  for (int c = 0; c < 6; ++c) row[3 + c] = pt[c];       // l0_points (6 feats)
  const float* q0 = fp2o + ((size_t)b * 512 + i0) * 128;
  const float* q1 = fp2o + ((size_t)b * 512 + i1) * 128;
  const float* q2 = fp2o + ((size_t)b * 512 + i2) * 128;
  for (int c = 0; c < 128; ++c)
    row[9 + c] = w0 * q0[c] + w1 * q1[c] + w2 * q2[c];
  for (int k = 137; k < 160; ++k) row[k] = 0.0f;
}

// ---------------------------------------------------------------------------
// log_softmax over the N axis of x^T: out[b,c,n] (one block per (b,c))
// ---------------------------------------------------------------------------
__global__ __launch_bounds__(256) void k_logsoftmax(const float* __restrict__ logits,
                                                    int ldl, int N, int C,
                                                    float* __restrict__ out) {
  int b = blockIdx.x / C, c = blockIdx.x % C, tid = threadIdx.x;
  const float* L = logits + (size_t)b * N * ldl + c;
  __shared__ float red[256];
  float m = -3.0e38f;
  for (int n = tid; n < N; n += 256) m = fmaxf(m, L[(size_t)n * ldl]);
  red[tid] = m; __syncthreads();
  for (int s = 128; s > 0; s >>= 1) {
    if (tid < s) red[tid] = fmaxf(red[tid], red[tid + s]);
    __syncthreads();
  }
  float mx = red[0];
  __syncthreads();
  float ssum = 0.0f;
  for (int n = tid; n < N; n += 256) ssum += expf(L[(size_t)n * ldl] - mx);
  red[tid] = ssum; __syncthreads();
  for (int s = 128; s > 0; s >>= 1) {
    if (tid < s) red[tid] += red[tid + s];
    __syncthreads();
  }
  float ls = logf(red[0]) + mx;
  float* O = out + ((size_t)b * C + c) * N;
  for (int n = tid; n < N; n += 256) O[n] = L[(size_t)n * ldl] - ls;
}

__global__ void k_copy(const float* __restrict__ src, float* __restrict__ dst, int total) {
  int t = blockIdx.x * blockDim.x + threadIdx.x;
  if (t < total) dst[t] = src[t];
}

// ===========================================================================
extern "C" void kernel_launch(void* const* d_in, const int* in_sizes, int n_in,
                              void* d_out, int out_size, void* d_ws, size_t ws_size,
                              hipStream_t stream) {
  (void)in_sizes; (void)n_in; (void)out_size; (void)ws_size;
  const int B = 8, N = 16384;
  const float* xyz = (const float*)d_in[0];

  struct Lyr { const float *W, *b, *g, *bt; };
  auto getL = [&](int base) {
    Lyr l;
    l.W  = (const float*)d_in[base + 0];
    l.b  = (const float*)d_in[base + 1];
    l.g  = (const float*)d_in[base + 2];
    l.bt = (const float*)d_in[base + 3];
    return l;
  };
  // d_in layout (insertion-order flatten): xyz, sa1[3], sa2[3], sa3[3],
  // fp3[2], fp2[2], fp1[3], head{W1,b1,g1,beta1,W2,b2}
  Lyr sa1[3] = { getL(1),  getL(5),  getL(9)  };
  Lyr sa2[3] = { getL(13), getL(17), getL(21) };
  Lyr sa3[3] = { getL(25), getL(29), getL(33) };
  Lyr fp3[2] = { getL(37), getL(41) };
  Lyr fp2[2] = { getL(45), getL(49) };
  Lyr fp1[3] = { getL(53), getL(57), getL(61) };
  const float* hW1 = (const float*)d_in[65];
  const float* hb1 = (const float*)d_in[66];
  const float* hg1 = (const float*)d_in[67];
  const float* hbt1= (const float*)d_in[68];
  const float* hW2 = (const float*)d_in[69];
  const float* hb2 = (const float*)d_in[70];

  // ---- workspace bump allocator ----
  char* wsb = (char*)d_ws;
  size_t off = 0;
  auto alloc = [&](size_t bytes) -> void* {
    void* p = wsb + off;
    off = (off + bytes + 255) & ~(size_t)255;
    return p;
  };
  int*   fid1  = (int*)  alloc((size_t)B * 512 * 4);
  int*   fid2  = (int*)  alloc((size_t)B * 128 * 4);
  int*   idx1  = (int*)  alloc((size_t)B * 512 * 64 * 4);
  int*   idx2  = (int*)  alloc((size_t)B * 128 * 64 * 4);
  float* l1xyz = (float*)alloc((size_t)B * 512 * 3 * 4);
  float* l2xyz = (float*)alloc((size_t)B * 128 * 3 * 4);
  float* l1pts = (float*)alloc((size_t)B * 512 * 128 * 4);
  float* l2pts = (float*)alloc((size_t)B * 128 * 256 * 4);
  float* l3pts = (float*)alloc((size_t)B * 1024 * 4);
  float* fp3o  = (float*)alloc((size_t)B * 128 * 256 * 4);
  float* fp2o  = (float*)alloc((size_t)B * 512 * 128 * 4);
  float* accum = (float*)alloc(2048 * 4);         // [sum(1024) | sumsq(1024)]
  float* bsum  = accum;
  float* bsq   = accum + 1024;

  auto grid1 = [](long long total) { return dim3((unsigned)((total + 255) / 256)); };

  // ---- pre-pad weights to f16 [CoutP, Kp] ----
  auto padw = [&](const float* W, int K, int Kp, int Cout) -> _Float16* {
    int CoutP = (Cout + 31) & ~31;
    _Float16* Wp = (_Float16*)alloc((size_t)CoutP * Kp * 2);
    int total = CoutP * Kp;
    k_pad_w<<<grid1(total), dim3(256), 0, stream>>>(W, K, Cout, Wp, Kp, total);
    return Wp;
  };
  _Float16* w_sa1[3] = { padw(sa1[0].W, 9,   32,  64),
                         padw(sa1[1].W, 64,  64,  64),
                         padw(sa1[2].W, 64,  64,  128) };
  _Float16* w_sa2[3] = { padw(sa2[0].W, 131, 160, 128),
                         padw(sa2[1].W, 128, 128, 128),
                         padw(sa2[2].W, 128, 128, 256) };
  _Float16* w_sa3[3] = { padw(sa3[0].W, 259, 288, 256),
                         padw(sa3[1].W, 256, 256, 512),
                         padw(sa3[2].W, 512, 512, 1024) };
  _Float16* w_fp3[2] = { padw(fp3[0].W, 1280, 1280, 256),
                         padw(fp3[1].W, 256,  256,  256) };
  _Float16* w_fp2[2] = { padw(fp2[0].W, 384, 384, 256),
                         padw(fp2[1].W, 256, 256, 128) };
  _Float16* w_fp1[3] = { padw(fp1[0].W, 137, 160, 128),
                         padw(fp1[1].W, 128, 128, 128),
                         padw(fp1[2].W, 128, 128, 128) };
  _Float16* w_h1 = padw(hW1, 128, 128, 128);
  _Float16* w_h2 = padw(hW2, 128, 128, 10);

  // ---- big ping-pong activation buffers ----
  float* bufA  = (float*)alloc((size_t)131072 * 160 * 4);   // max A-side use
  float* bufB  = (float*)alloc((size_t)262144 * 128 * 4);   // max B-side use

  auto gemm = [&](const float* X, int Kp, const _Float16* Wp, const float* bias,
                  float* Y, int ldy, int M, int Cout) {
    dim3 g((M + 63) / 64, (Cout + 31) / 32);
    k_gemm<<<g, dim3(128), 0, stream>>>(X, Wp, bias, Y, ldy, Kp, Cout);
  };
  auto bn = [&](float* Y, int M, int C, const float* g_, const float* bt_) {
    k_zero<<<dim3(8), dim3(256), 0, stream>>>(accum, 2048);
    long long total = (long long)M * C;
    k_bn_stats<<<dim3((M + 63) / 64), dim3(256), 0, stream>>>(Y, total, C, bsum, bsq);
    k_bn_apply<<<grid1(total), dim3(256), 0, stream>>>(Y, total, C, 1.0f / (float)M,
                                                       bsum, bsq, g_, bt_);
  };

  // ============================ SA1 =======================================
  const int M1 = B * 512 * 64;                              // 262144
  k_fps<<<dim3(B), dim3(512), 0, stream>>>(xyz, N, 6, 512, fid1);
  k_gather<<<grid1(B * 512), dim3(256), 0, stream>>>(xyz, 6, N, fid1, 512, l1xyz, B * 512);
  k_ball<<<grid1(B * 512), dim3(256), 0, stream>>>(xyz, N, 6, l1xyz, 512, 64, 0.04f, idx1, B * 512);
  k_group<<<grid1(M1), dim3(256), 0, stream>>>(xyz, 6, N, xyz, 6, l1xyz, idx1, 512, 64, bufA, 32, M1);
  gemm(bufA, 32, w_sa1[0], sa1[0].b, bufB, 64,  M1, 64);  bn(bufB, M1, 64,  sa1[0].g, sa1[0].bt);
  gemm(bufB, 64, w_sa1[1], sa1[1].b, bufA, 64,  M1, 64);  bn(bufA, M1, 64,  sa1[1].g, sa1[1].bt);
  gemm(bufA, 64, w_sa1[2], sa1[2].b, bufB, 128, M1, 128); bn(bufB, M1, 128, sa1[2].g, sa1[2].bt);
  k_maxpool<<<grid1((long long)B * 512 * 128), dim3(256), 0, stream>>>(bufB, 64, 128, l1pts, B * 512 * 128);

  // ============================ SA2 =======================================
  const int M2 = B * 128 * 64;                              // 65536
  k_fps<<<dim3(B), dim3(512), 0, stream>>>(l1xyz, 512, 3, 128, fid2);
  k_gather<<<grid1(B * 128), dim3(256), 0, stream>>>(l1xyz, 3, 512, fid2, 128, l2xyz, B * 128);
  k_ball<<<grid1(B * 128), dim3(256), 0, stream>>>(l1xyz, 512, 3, l2xyz, 128, 64, 0.16f, idx2, B * 128);
  k_group<<<grid1(M2), dim3(256), 0, stream>>>(l1xyz, 3, 512, l1pts, 128, l2xyz, idx2, 128, 64, bufA, 160, M2);
  gemm(bufA, 160, w_sa2[0], sa2[0].b, bufB, 128, M2, 128); bn(bufB, M2, 128, sa2[0].g, sa2[0].bt);
  gemm(bufB, 128, w_sa2[1], sa2[1].b, bufA, 128, M2, 128); bn(bufA, M2, 128, sa2[1].g, sa2[1].bt);
  gemm(bufA, 128, w_sa2[2], sa2[2].b, bufB, 256, M2, 256); bn(bufB, M2, 256, sa2[2].g, sa2[2].bt);
  k_maxpool<<<grid1((long long)B * 128 * 256), dim3(256), 0, stream>>>(bufB, 64, 256, l2pts, B * 128 * 256);

  // ============================ SA3 (group_all) ===========================
  const int M3 = B * 128;                                   // 1024
  k_group_all<<<grid1(M3), dim3(256), 0, stream>>>(l2xyz, l2pts, 256, bufA, 288, M3);
  gemm(bufA, 288, w_sa3[0], sa3[0].b, bufB, 256,  M3, 256);  bn(bufB, M3, 256,  sa3[0].g, sa3[0].bt);
  gemm(bufB, 256, w_sa3[1], sa3[1].b, bufA, 512,  M3, 512);  bn(bufA, M3, 512,  sa3[1].g, sa3[1].bt);
  gemm(bufA, 512, w_sa3[2], sa3[2].b, bufB, 1024, M3, 1024); bn(bufB, M3, 1024, sa3[2].g, sa3[2].bt);
  k_maxpool<<<grid1((long long)B * 1024), dim3(256), 0, stream>>>(bufB, 128, 1024, l3pts, B * 1024);

  // ============================ FP3 =======================================
  k_fp3_build<<<grid1(M3), dim3(256), 0, stream>>>(l2pts, l3pts, bufA, M3);
  gemm(bufA, 1280, w_fp3[0], fp3[0].b, bufB, 256, M3, 256); bn(bufB, M3, 256, fp3[0].g, fp3[0].bt);
  gemm(bufB, 256,  w_fp3[1], fp3[1].b, fp3o, 256, M3, 256); bn(fp3o, M3, 256, fp3[1].g, fp3[1].bt);

  // ============================ FP2 =======================================
  const int Mf2 = B * 512;                                  // 4096
  k_fp_build<<<grid1(Mf2), dim3(256), 0, stream>>>(l1xyz, l2xyz, 512, 128,
                                                   l1pts, 128, fp3o, 256, bufA, 384, Mf2);
  gemm(bufA, 384, w_fp2[0], fp2[0].b, bufB, 256, Mf2, 256); bn(bufB, Mf2, 256, fp2[0].g, fp2[0].bt);
  gemm(bufB, 256, w_fp2[1], fp2[1].b, fp2o, 128, Mf2, 128); bn(fp2o, Mf2, 128, fp2[1].g, fp2[1].bt);

  // ============================ FP1 =======================================
  const int Mf1 = B * N;                                    // 131072
  k_fp1_build<<<grid1(Mf1), dim3(256), 0, stream>>>(xyz, l1xyz, fp2o, bufA, Mf1);
  gemm(bufA, 160, w_fp1[0], fp1[0].b, bufB, 128, Mf1, 128); bn(bufB, Mf1, 128, fp1[0].g, fp1[0].bt);
  gemm(bufB, 128, w_fp1[1], fp1[1].b, bufA, 128, Mf1, 128); bn(bufA, Mf1, 128, fp1[1].g, fp1[1].bt);
  gemm(bufA, 128, w_fp1[2], fp1[2].b, bufB, 128, Mf1, 128); bn(bufB, Mf1, 128, fp1[2].g, fp1[2].bt);

  // ============================ Head ======================================
  gemm(bufB, 128, w_h1, hb1, bufA, 128, Mf1, 128); bn(bufA, Mf1, 128, hg1, hbt1);
  gemm(bufA, 128, w_h2, hb2, bufB, 16, Mf1, 10);   // logits [Mf1,16], 10 valid

  float* out = (float*)d_out;
  k_logsoftmax<<<dim3(B * 10), dim3(256), 0, stream>>>(bufB, 16, N, 10, out);
  k_copy<<<grid1(B * 1024), dim3(256), 0, stream>>>(l3pts, out + (size_t)B * 10 * N, B * 1024);
}
